// S4Checkpointed_48627619725769
// MI455X (gfx1250) — matrix operand order repaired and checked by reading, not compiled, not run on hardware
//
#include <hip/hip_runtime.h>
#include <hip/hip_bf16.h>

#define D_IN    256
#define D_HID   64
#define DT_RANK 16
#define D_MEM   128
#define STEP    16
#define BATCH   2
#define LSEQ    2048
#define BL      (BATCH * LSEQ)   // 4096 rows
#define D_OUTP  736              // 3*64 + 2*256 + 2*16
#define NMEM    (LSEQ / STEP)    // 128 memory slots per batch
#define NSCAN   (2 * BATCH * D_IN)  // 1024 independent scans
#define NCHUNK  16
#define CHUNK_L (LSEQ / NCHUNK)  // 128 steps per chunk
#define CHUNK_SHIFT 4

typedef __attribute__((ext_vector_type(16))) __bf16 v16bf;
typedef __attribute__((ext_vector_type(8)))  __bf16 v8bf;
typedef __attribute__((ext_vector_type(8)))  float  v8f;

// ---------------- WMMA helpers (layouts per CDNA5 ISA 7.12.2) ----------------

static __device__ __forceinline__ v8f wmma_bf16(v16bf a, v16bf b, v8f c) {
  // (neg_a, A, neg_b, B, c_mod, C, reuse_a, reuse_b)
  return __builtin_amdgcn_wmma_f32_16x16x32_bf16(false, a, false, b, (short)0, c,
                                                 false, false);
}

// A fragment: 16x32 bf16 tile, row-major source (leading dim `ld` elements).
static __device__ __forceinline__ v16bf load_a_frag(const __bf16* src, int ld,
                                                    int kb, int lane) {
  const int m = lane & 15, half = lane >> 4;
  const __bf16* p = src + (size_t)m * ld + kb + half * 8;
  v8bf lo = *(const v8bf*)(p);
  v8bf hi = *(const v8bf*)(p + 16);
  return __builtin_shufflevector(lo, hi, 0, 1, 2, 3, 4, 5, 6, 7,
                                 8, 9, 10, 11, 12, 13, 14, 15);
}

// B fragment: 32x16 bf16 tile from a transposed ("Bt[n][k]") source.
static __device__ __forceinline__ v16bf load_b_frag(const __bf16* bt, int ld,
                                                    int kb, int lane) {
  const int n = lane & 15, half = lane >> 4;
  const __bf16* p = bt + (size_t)n * ld + kb + half * 8;
  v8bf lo = *(const v8bf*)(p);
  v8bf hi = *(const v8bf*)(p + 16);
  return __builtin_shufflevector(lo, hi, 0, 1, 2, 3, 4, 5, 6, 7,
                                 8, 9, 10, 11, 12, 13, 14, 15);
}

static __device__ __forceinline__ float softplus_f(float v) {
  return (v > 20.0f) ? v : log1pf(__expf(v));
}

// ---------------- Kernel 1: bf16 casts + weight transposes ----------------

__global__ void prep_kernel(const float* __restrict__ x,
                            const float* __restrict__ wx,   // (256,736)
                            const float* __restrict__ wq,   // (256,128)
                            const float* __restrict__ wo,   // (128,256)
                            __bf16* __restrict__ xbf,       // (4096,256)
                            __bf16* __restrict__ wxt,       // (736,256)
                            __bf16* __restrict__ wqt,       // (128,256)
                            __bf16* __restrict__ wot) {     // (256,128)
  const int N1 = BL * D_IN;
  const int N2 = D_OUTP * D_IN;
  const int N3 = D_MEM * D_IN;
  const int N4 = D_IN * D_MEM;
  int t = blockIdx.x * blockDim.x + threadIdx.x;
  if (t < N1) {
    xbf[t] = (__bf16)x[t];
  } else if (t < N1 + N2) {
    int i = t - N1;
    int nn = i / D_IN, kk = i % D_IN;
    wxt[i] = (__bf16)wx[kk * D_OUTP + nn];
  } else if (t < N1 + N2 + N3) {
    int i = t - N1 - N2;
    int nn = i / D_IN, kk = i % D_IN;
    wqt[i] = (__bf16)wq[kk * D_MEM + nn];
  } else if (t < N1 + N2 + N3 + N4) {
    int i = t - N1 - N2 - N3;
    int dd = i / D_MEM, mm = i % D_MEM;
    wot[i] = (__bf16)wo[mm * D_IN + dd];
  }
}

// ---------------- Kernel 2: proj = x @ W_xproj (WMMA) ----------------

__global__ void proj_kernel(const __bf16* __restrict__ xbf,
                            const __bf16* __restrict__ wxt,
                            float* __restrict__ proj) {
  const int tileN = blockIdx.x;      // 0..45
  const int tileM = blockIdx.y;      // 0..255
  const int lane = threadIdx.x;
  const __bf16* a_src = xbf + (size_t)tileM * 16 * D_IN;
  const __bf16* b_src = wxt + (size_t)tileN * 16 * D_IN;
  v8f c = {};
#pragma unroll
  for (int kb = 0; kb < D_IN; kb += 32) {
    v16bf a = load_a_frag(a_src, D_IN, kb, lane);
    v16bf bfrag = load_b_frag(b_src, D_IN, kb, lane);
    c = wmma_bf16(a, bfrag, c);
  }
  const int n = lane & 15, half = lane >> 4;
  const int col = tileN * 16 + n;
#pragma unroll
  for (int r = 0; r < 8; r++) {
    int row = tileM * 16 + half * 8 + r;
    proj[(size_t)row * D_OUTP + col] = c[r];
  }
}

// ---------------- Kernel 3: delta projections (K=16, VALU) ----------------

__global__ void delta_kernel(const float* __restrict__ proj,
                             const float* __restrict__ Wdt,
                             const float* __restrict__ bdt,
                             const float* __restrict__ Wdtl,
                             const float* __restrict__ bdtl,
                             float* __restrict__ delta,
                             float* __restrict__ delta_l) {
  int t = blockIdx.x * blockDim.x + threadIdx.x;
  if (t >= BL * D_IN) return;
  int r = t >> 8, d = t & 255;
  const float* prow = proj + (size_t)r * D_OUTP;
  float acc = bdt[d], accl = bdtl[d];
#pragma unroll
  for (int j = 0; j < DT_RANK; j++) {
    acc  = fmaf(prow[704 + j], Wdt[j * D_IN + d], acc);
    accl = fmaf(prow[720 + j], Wdtl[j * D_IN + d], accl);
  }
  delta[t]   = softplus_f(acc);
  delta_l[t] = softplus_f(accl);
}

// ---------------- Chunked parallel scan ----------------
// scan id sid in [0,1024): variant = sid>>9, b = (sid>>8)&1, d = sid&255.
// Lane holds states n0=lane, n1=lane+32 (D_HID=64 across wave32 x2).
// Recurrence per step: s = exp(delta*A)*s + delta*B*u.
//
// Pass 1: per (sid, chunk) compute chunk transition prodA = exp(sum delta*A)
//         and chunk-final state (zero init).
// Pass 2: per sid, 16-step sequential combine -> per-chunk initial states.
// Pass 3: per (sid, chunk) re-run with correct init, emit y / memory.

__global__ void scan_part1(const float* __restrict__ x,
                           const float* __restrict__ proj,
                           const float* __restrict__ delta,
                           const float* __restrict__ delta_l,
                           const float* __restrict__ A,
                           float* __restrict__ pA,
                           float* __restrict__ fin) {
  const int c   = blockIdx.x & (NCHUNK - 1);
  const int sid = blockIdx.x >> CHUNK_SHIFT;
  const int variant = sid >> 9;
  const int b = (sid >> 8) & 1;
  const int d = sid & 255;
  const int lane = threadIdx.x;
  const int n0 = lane, n1 = lane + 32;

  const float a0 = A[d * D_HID + n0];
  const float a1 = A[d * D_HID + n1];
  const float* del = (variant == 0) ? delta : delta_l;
  const int bbase = (variant == 0) ? 0 : 64;

  float t0 = 0.0f, t1 = 0.0f;   // sum of delta*A (log of prodA)
  float s0 = 0.0f, s1 = 0.0f;
  const int l0 = c * CHUNK_L;
  for (int l = l0; l < l0 + CHUNK_L; l++) {
    const int r = b * LSEQ + l;
    const float* prow = proj + (size_t)r * D_OUTP;
    if (l + 2 < l0 + CHUNK_L) {
      __builtin_prefetch(prow + 2 * D_OUTP, 0, 0);
      __builtin_prefetch(del + (size_t)(r + 2) * D_IN + d, 0, 0);
    }
    const float dv = del[(size_t)r * D_IN + d];
    const float u  = x[(size_t)r * D_IN + d];
    const float du = dv * u;
    const float dva0 = dv * a0, dva1 = dv * a1;
    t0 += dva0;
    t1 += dva1;
    s0 = fmaf(__expf(dva0), s0, du * prow[bbase + n0]);
    s1 = fmaf(__expf(dva1), s1, du * prow[bbase + n1]);
  }
  const size_t base = ((size_t)sid * NCHUNK + c) * D_HID;
  pA[base + n0] = __expf(t0);
  pA[base + n1] = __expf(t1);
  fin[base + n0] = s0;
  fin[base + n1] = s1;
}

__global__ void scan_combine(const float* __restrict__ pA,
                             const float* __restrict__ fin,
                             float* __restrict__ initst) {
  const int sid = blockIdx.x;   // 0..1023
  const int lane = threadIdx.x;
  const int n0 = lane, n1 = lane + 32;
  float i0 = 0.0f, i1 = 0.0f;
  for (int c = 0; c < NCHUNK; c++) {
    const size_t base = ((size_t)sid * NCHUNK + c) * D_HID;
    initst[base + n0] = i0;
    initst[base + n1] = i1;
    i0 = fmaf(pA[base + n0], i0, fin[base + n0]);
    i1 = fmaf(pA[base + n1], i1, fin[base + n1]);
  }
}

__global__ void scan_part2(const float* __restrict__ x,
                           const float* __restrict__ proj,
                           const float* __restrict__ delta,
                           const float* __restrict__ delta_l,
                           const float* __restrict__ A,
                           const float* __restrict__ Dp,
                           const float* __restrict__ initst,
                           float* __restrict__ yv,
                           float* __restrict__ memv) {
  const int c   = blockIdx.x & (NCHUNK - 1);
  const int sid = blockIdx.x >> CHUNK_SHIFT;
  const int variant = sid >> 9;
  const int b = (sid >> 8) & 1;
  const int d = sid & 255;
  const int lane = threadIdx.x;
  const int n0 = lane, n1 = lane + 32;

  const float a0 = A[d * D_HID + n0];
  const float a1 = A[d * D_HID + n1];
  const float dp = Dp[d];
  const float* del = (variant == 0) ? delta : delta_l;
  const int bbase = (variant == 0) ? 0 : 64;

  const size_t base = ((size_t)sid * NCHUNK + c) * D_HID;
  float s0 = initst[base + n0];
  float s1 = initst[base + n1];
  const int l0 = c * CHUNK_L;
  for (int l = l0; l < l0 + CHUNK_L; l++) {
    const int r = b * LSEQ + l;
    const float* prow = proj + (size_t)r * D_OUTP;
    if (l + 2 < l0 + CHUNK_L) {
      __builtin_prefetch(prow + 2 * D_OUTP, 0, 0);
      __builtin_prefetch(del + (size_t)(r + 2) * D_IN + d, 0, 0);
    }
    const float dv = del[(size_t)r * D_IN + d];
    const float u  = x[(size_t)r * D_IN + d];
    const float du = dv * u;
    s0 = fmaf(__expf(dv * a0), s0, du * prow[bbase + n0]);
    s1 = fmaf(__expf(dv * a1), s1, du * prow[bbase + n1]);
    if (variant == 0) {
      float v = fmaf(s0, prow[128 + n0], s1 * prow[128 + n1]);
#pragma unroll
      for (int o = 16; o > 0; o >>= 1) v += __shfl_xor(v, o, 32);
      if (lane == 0) yv[(size_t)r * D_IN + d] = v + u * dp;
    } else if ((l & (STEP - 1)) == 0) {
      float v = s0 + s1;
#pragma unroll
      for (int o = 16; o > 0; o >>= 1) v += __shfl_xor(v, o, 32);
      if (lane == 0) memv[((size_t)b * NMEM + (l >> 4)) * D_IN + d] = v;
    }
  }
}

// ---------------- Kernel 5: K/V projections from memory ----------------

__global__ void kv_kernel(const float* __restrict__ memv,
                          const float* __restrict__ WK,
                          const float* __restrict__ bK,
                          const float* __restrict__ WV,
                          const float* __restrict__ bV,
                          __bf16* __restrict__ kbf,
                          __bf16* __restrict__ vtbf) {
  int t = blockIdx.x * blockDim.x + threadIdx.x;
  if (t >= BATCH * NMEM * D_MEM) return;
  int m = t & 127;
  int k = (t >> 7) & 127;
  int b = t >> 14;
  const float* mrow = memv + ((size_t)b * NMEM + k) * D_IN;
  float ak = bK[m], av = bV[m];
  for (int j = 0; j < D_IN; j++) {
    float mv = mrow[j];
    ak = fmaf(mv, WK[j * D_MEM + m], ak);
    av = fmaf(mv, WV[j * D_MEM + m], av);
  }
  kbf[((size_t)b * NMEM + k) * D_MEM + m]  = (__bf16)ak;
  vtbf[((size_t)b * D_MEM + m) * NMEM + k] = (__bf16)av;
}

// ---------------- Kernel 6: fused attention + epilogue ----------------

__global__ void attn_kernel(const __bf16* __restrict__ xbf,
                            const __bf16* __restrict__ wqt,
                            const float* __restrict__ bQ,
                            const __bf16* __restrict__ kbf,
                            const __bf16* __restrict__ vtbf,
                            const __bf16* __restrict__ wot,
                            const float* __restrict__ bO,
                            const float* __restrict__ proj,
                            const float* __restrict__ x,
                            const float* __restrict__ yv,
                            float* __restrict__ out) {
  __shared__ __align__(32) __bf16 qbuf[16 * D_MEM];
  __shared__ __align__(32) float  sbuf[16 * D_MEM];
  __shared__ __align__(32) __bf16 wbuf[16 * D_MEM];
  __shared__ __align__(32) __bf16 abuf[16 * D_MEM];

  const int tile = blockIdx.x;          // 0..255
  const int lane = threadIdx.x;
  const int n = lane & 15, half = lane >> 4;
  const int row0 = tile * 16;
  const int b = row0 >> 11;             // 2048 rows per batch
  const __bf16* a_src = xbf + (size_t)row0 * D_IN;

  // Q = x @ W_Q + b_Q  (16 x 128)
#pragma unroll
  for (int nb = 0; nb < 8; nb++) {
    v8f c = {};
#pragma unroll
    for (int kb = 0; kb < D_IN; kb += 32) {
      v16bf a = load_a_frag(a_src, D_IN, kb, lane);
      v16bf bfrag = load_b_frag(wqt + (size_t)nb * 16 * D_IN, D_IN, kb, lane);
      c = wmma_bf16(a, bfrag, c);
    }
    int col = nb * 16 + n;
    float bias = bQ[col];
#pragma unroll
    for (int r = 0; r < 8; r++)
      qbuf[(half * 8 + r) * D_MEM + col] = (__bf16)(c[r] + bias);
  }
  __syncthreads();

  // scores = Q @ K^T * (1/sqrt(64))  (16 x 128 keys)
  const __bf16* k_src = kbf + (size_t)b * NMEM * D_MEM;
#pragma unroll
  for (int nb = 0; nb < 8; nb++) {
    v8f c = {};
#pragma unroll
    for (int kb = 0; kb < D_MEM; kb += 32) {
      v16bf a = load_a_frag(qbuf, D_MEM, kb, lane);
      v16bf bfrag = load_b_frag(k_src + (size_t)nb * 16 * D_MEM, D_MEM, kb, lane);
      c = wmma_bf16(a, bfrag, c);
    }
    int col = nb * 16 + n;
#pragma unroll
    for (int r = 0; r < 8; r++)
      sbuf[(half * 8 + r) * D_MEM + col] = c[r] * 0.125f;
  }
  __syncthreads();

  // row softmax over 128 keys (lanes 0..15 own one row each)
  if (lane < 16) {
    float mx = -3.4e38f;
    for (int j = 0; j < D_MEM; j++) mx = fmaxf(mx, sbuf[lane * D_MEM + j]);
    float sum = 0.0f;
    for (int j = 0; j < D_MEM; j++) {
      float e = __expf(sbuf[lane * D_MEM + j] - mx);
      sbuf[lane * D_MEM + j] = e;
      sum += e;
    }
    float inv = 1.0f / sum;
    for (int j = 0; j < D_MEM; j++)
      wbuf[lane * D_MEM + j] = (__bf16)(sbuf[lane * D_MEM + j] * inv);
  }
  __syncthreads();

  // att = w @ V  (16 x 128), contraction over 128 keys; B = V^T[b][m][k]
  const __bf16* v_src = vtbf + (size_t)b * D_MEM * NMEM;
#pragma unroll
  for (int nb = 0; nb < 8; nb++) {
    v8f c = {};
#pragma unroll
    for (int kb = 0; kb < NMEM; kb += 32) {
      v16bf a = load_a_frag(wbuf, NMEM, kb, lane);
      v16bf bfrag = load_b_frag(v_src + (size_t)nb * 16 * NMEM, NMEM, kb, lane);
      c = wmma_bf16(a, bfrag, c);
    }
    int col = nb * 16 + n;
#pragma unroll
    for (int r = 0; r < 8; r++)
      abuf[(half * 8 + r) * D_MEM + col] = (__bf16)c[r];
  }
  __syncthreads();

  // out = y + (att @ W_O + b_O) * E + x * Fg    (16 x 256)
#pragma unroll
  for (int nb = 0; nb < 16; nb++) {
    v8f c = {};
#pragma unroll
    for (int kb = 0; kb < D_MEM; kb += 32) {
      v16bf a = load_a_frag(abuf, D_MEM, kb, lane);
      v16bf bfrag = load_b_frag(wot + (size_t)nb * 16 * D_MEM, D_MEM, kb, lane);
      c = wmma_bf16(a, bfrag, c);
    }
    int col = nb * 16 + n;
    float bias = bO[col];
#pragma unroll
    for (int r = 0; r < 8; r++) {
      int row = row0 + half * 8 + r;
      float att = c[r] + bias;
      float E  = proj[(size_t)row * D_OUTP + 192 + col];
      float Fg = proj[(size_t)row * D_OUTP + 448 + col];
      out[(size_t)row * D_IN + col] =
          yv[(size_t)row * D_IN + col] + att * E +
          x[(size_t)row * D_IN + col] * Fg;
    }
  }
}

// ---------------- host launcher ----------------

extern "C" void kernel_launch(void* const* d_in, const int* in_sizes, int n_in,
                              void* d_out, int out_size, void* d_ws, size_t ws_size,
                              hipStream_t stream) {
  (void)in_sizes; (void)n_in; (void)out_size; (void)ws_size;
  const float* x    = (const float*)d_in[0];
  const float* Wx   = (const float*)d_in[1];
  const float* Wdt  = (const float*)d_in[2];
  const float* bdt  = (const float*)d_in[3];
  const float* Wdtl = (const float*)d_in[4];
  const float* bdtl = (const float*)d_in[5];
  const float* A    = (const float*)d_in[6];
  const float* Dp   = (const float*)d_in[7];
  const float* WQ   = (const float*)d_in[8];
  const float* bQ   = (const float*)d_in[9];
  const float* WK   = (const float*)d_in[10];
  const float* bK   = (const float*)d_in[11];
  const float* WV   = (const float*)d_in[12];
  const float* bV   = (const float*)d_in[13];
  const float* WO   = (const float*)d_in[14];
  const float* bO   = (const float*)d_in[15];
  float* out = (float*)d_out;

  char* ws = (char*)d_ws;
  size_t off = 0;
  auto alloc = [&](size_t bytes) -> void* {
    void* p = ws + off;
    off += (bytes + 255) & ~(size_t)255;
    return p;
  };
  float*  proj    = (float*)alloc(sizeof(float) * BL * D_OUTP);
  float*  delta   = (float*)alloc(sizeof(float) * BL * D_IN);
  float*  delta_l = (float*)alloc(sizeof(float) * BL * D_IN);
  float*  yv      = (float*)alloc(sizeof(float) * BL * D_IN);
  float*  memv    = (float*)alloc(sizeof(float) * BATCH * NMEM * D_IN);
  float*  pA      = (float*)alloc(sizeof(float) * NSCAN * NCHUNK * D_HID);
  float*  fin     = (float*)alloc(sizeof(float) * NSCAN * NCHUNK * D_HID);
  float*  initst  = (float*)alloc(sizeof(float) * NSCAN * NCHUNK * D_HID);
  __bf16* xbf     = (__bf16*)alloc(sizeof(__bf16) * BL * D_IN);
  __bf16* wxt     = (__bf16*)alloc(sizeof(__bf16) * D_OUTP * D_IN);
  __bf16* wqt     = (__bf16*)alloc(sizeof(__bf16) * D_MEM * D_IN);
  __bf16* wot     = (__bf16*)alloc(sizeof(__bf16) * D_IN * D_MEM);
  __bf16* kbf     = (__bf16*)alloc(sizeof(__bf16) * BATCH * NMEM * D_MEM);
  __bf16* vtbf    = (__bf16*)alloc(sizeof(__bf16) * BATCH * D_MEM * NMEM);

  {
    int total = BL * D_IN + D_OUTP * D_IN + D_MEM * D_IN + D_IN * D_MEM;
    prep_kernel<<<(total + 255) / 256, 256, 0, stream>>>(x, Wx, WQ, WO,
                                                         xbf, wxt, wqt, wot);
  }
  proj_kernel<<<dim3(D_OUTP / 16, BL / 16), 32, 0, stream>>>(xbf, wxt, proj);
  delta_kernel<<<(BL * D_IN + 255) / 256, 256, 0, stream>>>(proj, Wdt, bdt,
                                                            Wdtl, bdtl,
                                                            delta, delta_l);
  scan_part1<<<NSCAN * NCHUNK, 32, 0, stream>>>(x, proj, delta, delta_l, A,
                                                pA, fin);
  scan_combine<<<NSCAN, 32, 0, stream>>>(pA, fin, initst);
  scan_part2<<<NSCAN * NCHUNK, 32, 0, stream>>>(x, proj, delta, delta_l, A, Dp,
                                                initst, yv, memv);
  kv_kernel<<<(BATCH * NMEM * D_MEM + 255) / 256, 256, 0, stream>>>(
      memv, WK, bK, WV, bV, kbf, vtbf);
  attn_kernel<<<BL / 16, 32, 0, stream>>>(xbf, wqt, bQ, kbf, vtbf, wot, bO,
                                          proj, x, yv, out);
}
